// pairnormRGCN_43817256354377
// MI455X (gfx1250) — matrix operand order repaired and checked by reading, not compiled, not run on hardware
//
#include <hip/hip_runtime.h>

// ---------------- CDNA5 (gfx1250) RGCN + PairNorm pipeline -----------------
// Strategy (memory-bound problem, 23.3 TB/s HBM, 192MB L2):
//   1. Pack all weight matrices (Wproj, root1/2, basis-combined W1/W2) into
//      bf16 B-fragment layout for v_wmma_f32_16x16x32_bf16.
//   2. h0 = relu(x @ Wproj + b)          -> fp32 (output) + bf16 copy
//   3. T1[n, r*128+o] = h0 @ W1_r        -> bf16 (205MB, gathered per edge)
//      agg = h0 @ root1 + bias1          -> fp32 (51MB, L2-resident)
//      per edge: agg[dst] += T1[src,et]/cnt[dst,et]   (f32 atomics into L2)
//      h1 = pairnorm(relu(agg))
//   4. same for layer 2 with Fout=64, no relu, -> final output.
// GEMM is double-buffered on B fragments so the next tile's global_load_b128
// clause overlaps the current tile's 4 WMMAs; all epilogue variants are
// compile-time template parameters (no branches in the hot loop).

#define HIDF 128
#define KDIM 128

typedef __attribute__((ext_vector_type(16))) __bf16          v16bf;
typedef __attribute__((ext_vector_type(8)))  float           v8f;
typedef __attribute__((ext_vector_type(4)))  unsigned int    v4u;
typedef __attribute__((ext_vector_type(4)))  unsigned short  v4s;
typedef __attribute__((ext_vector_type(2)))  unsigned short  v2s;

union FragB { v16bf v; v4u q[2]; };

__device__ __forceinline__ unsigned short f2b(float f) {
    union { float f; unsigned int u; } x; x.f = f;
    unsigned int r = x.u + 0x7FFFu + ((x.u >> 16) & 1u);   // round-nearest-even
    return (unsigned short)(r >> 16);
}
__device__ __forceinline__ float b2f(unsigned short h) {
    union { unsigned int u; float f; } x; x.u = ((unsigned int)h) << 16; return x.f;
}

// ---- pack a plain [K x Ncols] fp32 matrix into bf16 WMMA B-fragment layout
// packed index = ((nt*(K/32) + s)*32 + lane)*16 + i ; element (v=i>>1, odd=i&1)
// maps to  k = s*32 + (lane>=16 ? 16 : 0) + 2v + odd ,  n = nt*16 + (lane&15)
__global__ void pack_plain_kernel(const float* __restrict__ src,
                                  unsigned short* __restrict__ dst,
                                  int K, int Ncols) {
    int total = K * Ncols;
    int nslabs = K >> 5;
    for (int idx = blockIdx.x * blockDim.x + threadIdx.x; idx < total;
         idx += gridDim.x * blockDim.x) {
        int i    = idx & 15;
        int lane = (idx >> 4) & 31;
        int slab = idx >> 9;
        int s  = slab % nslabs;
        int nt = slab / nslabs;
        int k = s * 32 + ((lane >= 16) ? 16 : 0) + 2 * (i >> 1) + (i & 1);
        int n = nt * 16 + (lane & 15);
        dst[idx] = f2b(src[(size_t)k * Ncols + n]);
    }
}

// ---- combine basis: W_r = sum_b comp[r,b]*basis[b], pack [K x (R*Fout)]
__global__ void pack_basis_kernel(const float* __restrict__ basis,
                                  const float* __restrict__ comp,
                                  unsigned short* __restrict__ dst,
                                  int K, int Fout, int R, int NB) {
    int Ncols = R * Fout;
    int total = K * Ncols;
    int nslabs = K >> 5;
    for (int idx = blockIdx.x * blockDim.x + threadIdx.x; idx < total;
         idx += gridDim.x * blockDim.x) {
        int i    = idx & 15;
        int lane = (idx >> 4) & 31;
        int slab = idx >> 9;
        int s  = slab % nslabs;
        int nt = slab / nslabs;
        int k = s * 32 + ((lane >= 16) ? 16 : 0) + 2 * (i >> 1) + (i & 1);
        int n = nt * 16 + (lane & 15);
        int r = n / Fout, fo = n % Fout;
        float acc = 0.f;
        for (int b = 0; b < NB; ++b)
            acc += comp[r * NB + b] * basis[((size_t)b * K + k) * Fout + fo];
        dst[idx] = f2b(acc);
    }
}

__global__ void cvt_bf16_kernel(const float* __restrict__ src,
                                unsigned short* __restrict__ dst, size_t total) {
    for (size_t i = blockIdx.x * (size_t)blockDim.x + threadIdx.x; i < total;
         i += (size_t)gridDim.x * blockDim.x)
        dst[i] = f2b(src[i]);
}

__global__ void edge_count_kernel(const int* __restrict__ dst_idx,
                                  const int* __restrict__ etype,
                                  int* __restrict__ cnt, int E, int R) {
    for (int e = blockIdx.x * blockDim.x + threadIdx.x; e < E;
         e += gridDim.x * blockDim.x)
        atomicAdd(&cnt[(size_t)dst_idx[e] * R + etype[e]], 1);
}

__global__ void invcnt_kernel(float* buf, int total) {  // in-place int -> 1/max(c,1)
    for (int i = blockIdx.x * blockDim.x + threadIdx.x; i < total;
         i += gridDim.x * blockDim.x) {
        int c = ((const int*)buf)[i];
        buf[i] = (c > 0) ? (1.0f / (float)c) : 1.0f;
    }
}

__device__ __forceinline__ void load_bfrag(FragB* b,
                                           const unsigned short* __restrict__ Bpack,
                                           int lane, int nt) {
#pragma unroll
    for (int s = 0; s < 4; ++s) {
        const unsigned short* bp = Bpack + (((size_t)nt * 4 + s) * 32 + lane) * 16;
        b[s].q[0] = *(const v4u*)(bp);
        b[s].q[1] = *(const v4u*)(bp + 8);
    }
}

// ---- WMMA GEMM: C[M,Ncols] = act(A[M,128]bf16 @ Bpack + bias)
// One wave per 16-row tile; A fragments (4 k-slabs) held in regs across all
// n-tiles.  B fragments double-buffered: next tile's loads overlap current
// tile's 4 WMMAs.  Epilogue fully compile-time specialized.
template <bool HAS_BIAS, bool RELU, bool WF, bool WB>
__global__ void __launch_bounds__(256)
wmma_gemm_kernel(const unsigned short* __restrict__ A,
                 const unsigned short* __restrict__ Bpack,
                 const float* __restrict__ bias,
                 float* __restrict__ outF,
                 unsigned short* __restrict__ outB,
                 int M, int Ncols) {
    const int lane  = threadIdx.x & 31;
    const int wv    = threadIdx.x >> 5;
    const int mtile = blockIdx.x * 8 + wv;
    if (mtile * 16 >= M) return;                 // wave-uniform: EXEC stays all-1s
    const int halfsel = (lane >= 16) ? 1 : 0;
    const int mrow    = mtile * 16 + (lane & 15);
    const int kb      = halfsel * 16;

    FragB a[4];
    const unsigned short* arow = A + (size_t)mrow * KDIM;
#pragma unroll
    for (int s = 0; s < 4; ++s) {
        a[s].q[0] = *(const v4u*)(arow + s * 32 + kb);
        a[s].q[1] = *(const v4u*)(arow + s * 32 + 8 + kb);
    }

    const int ntiles = Ncols >> 4;
    FragB bcur[4], bnxt[4];
    load_bfrag(bcur, Bpack, lane, 0);

    const int col0  = (lane & 15);
    const int mbase = mtile * 16 + halfsel * 8;

    for (int nt = 0; nt < ntiles; ++nt) {
        if (nt + 1 < ntiles) load_bfrag(bnxt, Bpack, lane, nt + 1);

        v8f c = {};
#pragma unroll
        for (int s = 0; s < 4; ++s)
            c = __builtin_amdgcn_wmma_f32_16x16x32_bf16(
                    false, a[s].v, false, bcur[s].v, (short)0, c, false, false);

        const int cc = nt * 16 + col0;
        float bv = 0.0f;
        if (HAS_BIAS) bv = bias[cc];
#pragma unroll
        for (int j = 0; j < 8; ++j) {
            float v = c[j] + bv;
            if (RELU) v = fmaxf(v, 0.0f);
            if (WF) outF[(size_t)(mbase + j) * Ncols + cc] = v;
            if (WB) outB[(size_t)(mbase + j) * Ncols + cc] = f2b(v);
        }
#pragma unroll
        for (int s = 0; s < 4; ++s) bcur[s] = bnxt[s];
    }
}

// ---- per-edge scatter: agg[dst] += T[src,etype,:] * invcnt[dst,etype]
// One wave per edge; vector bf16 loads, f32 atomics into L2-resident agg.
template <int F>
__global__ void __launch_bounds__(256)
scatter_kernel(const unsigned short* __restrict__ T,
               const float* __restrict__ invcnt,
               const int* __restrict__ src, const int* __restrict__ dst,
               const int* __restrict__ et, float* __restrict__ agg,
               int E, int R) {
    const int lane = threadIdx.x & 31;
    const int e    = (int)((blockIdx.x * (size_t)blockDim.x + threadIdx.x) >> 5);
    if (e >= E) return;
    const int s = src[e], d = dst[e], t = et[e];
    const float ic = invcnt[(size_t)d * R + t];
    const unsigned short* row = T + ((size_t)s * R + t) * F;
    float* out = agg + (size_t)d * F;
    constexpr int EPL = F / 32;
    if (EPL == 4) {
        v4s p = *(const v4s*)(row + lane * 4);
#pragma unroll
        for (int i = 0; i < 4; ++i)
            atomicAdd(out + lane * 4 + i, b2f(p[i]) * ic);
    } else {
        v2s p = *(const v2s*)(row + lane * 2);
#pragma unroll
        for (int i = 0; i < 2; ++i)
            atomicAdd(out + lane * 2 + i, b2f(p[i]) * ic);
    }
}

// ---- PairNorm pass 1: column sums + total sum of squares (of act(X))
__global__ void __launch_bounds__(256)
stats_kernel(const float* __restrict__ X, float* __restrict__ stats,
             int Nn, int F, int doRelu) {
    __shared__ float scol[128];
    __shared__ float ssq;
    const int tid  = threadIdx.x;
    const int col  = tid % F;
    const int rseg = tid / F;
    const int rpi  = 256 / F;
    if (tid < F) scol[tid] = 0.f;
    if (tid == 0) ssq = 0.f;
    __syncthreads();
    float cs = 0.f, sq = 0.f;
    for (int row = blockIdx.x * rpi + rseg; row < Nn; row += gridDim.x * rpi) {
        float v = X[(size_t)row * F + col];
        if (doRelu) v = fmaxf(v, 0.f);
        cs += v; sq += v * v;
    }
    atomicAdd(&scol[col], cs);   // ds_add_f32
    atomicAdd(&ssq, sq);
    __syncthreads();
    if (tid < F)  atomicAdd(&stats[tid], scol[tid]);
    if (tid == 0) atomicAdd(&stats[F], ssq);
}

// ---- PairNorm finalize: stats[c] -> mu_c ; stats[F] -> 1/sqrt(eps + mean||x-mu||^2)
__global__ void finalize_kernel(float* stats, int Nn, int F) {
    __shared__ float red[128];
    const int t = threadIdx.x;
    float mu = 0.f;
    if (t < F) { mu = stats[t] / (float)Nn; stats[t] = mu; }
    red[t] = mu * mu;
    __syncthreads();
    for (int s = 64; s > 0; s >>= 1) {
        if (t < s) red[t] += red[t + s];
        __syncthreads();
    }
    if (t == 0) {
        float sumsq    = stats[F];
        float centered = sumsq - red[0] * (float)Nn;  // sum||x||^2 - N*||mu||^2
        stats[F] = rsqrtf(1e-5f + centered / (float)Nn);
    }
}

__global__ void normalize_kernel(const float* __restrict__ X,
                                 const float* __restrict__ stats,
                                 float* __restrict__ outF,
                                 unsigned short* __restrict__ outB,
                                 int Nn, int F, int doRelu) {
    const size_t total = (size_t)Nn * F;
    const float rinv = stats[F];
    for (size_t i = blockIdx.x * (size_t)blockDim.x + threadIdx.x; i < total;
         i += (size_t)gridDim.x * blockDim.x) {
        int c = (int)(i % F);
        float v = X[i];
        if (doRelu) v = fmaxf(v, 0.f);
        float o = (v - stats[c]) * rinv;
        outF[i] = o;
        if (outB) outB[i] = f2b(o);
    }
}

extern "C" void kernel_launch(void* const* d_in, const int* in_sizes, int n_in,
                              void* d_out, int out_size, void* d_ws, size_t ws_size,
                              hipStream_t stream) {
    (void)n_in; (void)out_size; (void)ws_size;
    const float* x      = (const float*)d_in[0];
    const float* Wproj  = (const float*)d_in[1];
    const float* bproj  = (const float*)d_in[2];
    const float* basis1 = (const float*)d_in[3];
    const float* comp1  = (const float*)d_in[4];
    const float* root1  = (const float*)d_in[5];
    const float* bias1  = (const float*)d_in[6];
    const float* basis2 = (const float*)d_in[7];
    const float* comp2  = (const float*)d_in[8];
    const float* root2  = (const float*)d_in[9];
    const float* bias2  = (const float*)d_in[10];
    const int*   eidx   = (const int*)d_in[11];
    const int*   etype  = (const int*)d_in[12];

    const int N  = in_sizes[0] / HIDF;    // 100000
    const int E  = in_sizes[12];          // 1600000
    const int R  = 8, NB = 4, OUTF = 64;
    const int* esrc = eidx;
    const int* edst = eidx + E;

    // ---- workspace carve-out (256B aligned)
    char* ws = (char*)d_ws;
    size_t off = 0;
    auto carve = [&](size_t bytes) -> char* {
        char* p = ws + off;
        off = (off + bytes + 255) & ~(size_t)255;
        return p;
    };
    float*          cntbuf = (float*)carve((size_t)N * R * 4);     // int counts -> inv counts
    float*          stats  = (float*)carve(256 * 4);
    unsigned short* xb     = (unsigned short*)carve((size_t)N * HIDF * 2);
    unsigned short* h0b    = (unsigned short*)carve((size_t)N * HIDF * 2);
    unsigned short* h1b    = (unsigned short*)carve((size_t)N * HIDF * 2);
    unsigned short* WprojP = (unsigned short*)carve((size_t)HIDF * HIDF * 2);
    unsigned short* root1P = (unsigned short*)carve((size_t)HIDF * HIDF * 2);
    unsigned short* root2P = (unsigned short*)carve((size_t)HIDF * OUTF * 2);
    unsigned short* W1P    = (unsigned short*)carve((size_t)HIDF * R * HIDF * 2);
    unsigned short* W2P    = (unsigned short*)carve((size_t)HIDF * R * OUTF * 2);
    unsigned short* Tbuf   = (unsigned short*)carve((size_t)N * R * HIDF * 2);
    float*          agg    = (float*)carve((size_t)N * HIDF * 4);

    float* outv      = (float*)d_out;
    float* out_final = outv;                         // [N,64]
    float* out_h0    = outv + (size_t)N * OUTF;      // [N,128]
    float* out_h1    = out_h0 + (size_t)N * HIDF;    // [N,128]

    hipMemsetAsync(cntbuf, 0, (size_t)N * R * 4, stream);
    hipMemsetAsync(stats, 0, 256 * 4, stream);

    // weight prep + fp32->bf16 conversion
    pack_plain_kernel<<<64, 256, 0, stream>>>(Wproj, WprojP, HIDF, HIDF);
    pack_plain_kernel<<<64, 256, 0, stream>>>(root1, root1P, HIDF, HIDF);
    pack_plain_kernel<<<32, 256, 0, stream>>>(root2, root2P, HIDF, OUTF);
    pack_basis_kernel<<<512, 256, 0, stream>>>(basis1, comp1, W1P, HIDF, HIDF, R, NB);
    pack_basis_kernel<<<256, 256, 0, stream>>>(basis2, comp2, W2P, HIDF, OUTF, R, NB);
    cvt_bf16_kernel<<<2048, 256, 0, stream>>>(x, xb, (size_t)N * HIDF);

    // edge counts -> inverse counts (L2-resident)
    edge_count_kernel<<<4096, 256, 0, stream>>>(edst, etype, (int*)cntbuf, E, R);
    invcnt_kernel<<<2048, 256, 0, stream>>>(cntbuf, N * R);

    const int mtiles  = (N + 15) / 16;
    const int gblocks = (mtiles + 7) / 8;
    const int eblocks = (E + 7) / 8;

    // ---- front-end: h0 = relu(x @ Wproj + b)   (bias, relu, fp32 out + bf16 copy)
    wmma_gemm_kernel<true, true, true, true>
        <<<gblocks, 256, 0, stream>>>(xb, WprojP, bproj, out_h0, h0b, N, HIDF);
    // ---- layer 1
    wmma_gemm_kernel<false, false, false, true>
        <<<gblocks, 256, 0, stream>>>(h0b, W1P, nullptr, nullptr, Tbuf, N, R * HIDF);
    wmma_gemm_kernel<true, false, true, false>
        <<<gblocks, 256, 0, stream>>>(h0b, root1P, bias1, agg, nullptr, N, HIDF);
    scatter_kernel<128><<<eblocks, 256, 0, stream>>>(Tbuf, cntbuf, esrc, edst,
                                                     etype, agg, E, R);
    stats_kernel<<<2048, 256, 0, stream>>>(agg, stats, N, HIDF, 1);
    finalize_kernel<<<1, 128, 0, stream>>>(stats, N, HIDF);
    normalize_kernel<<<4096, 256, 0, stream>>>(agg, stats, out_h1, h1b, N, HIDF, 1);

    // ---- layer 2
    hipMemsetAsync(stats, 0, 256 * 4, stream);
    wmma_gemm_kernel<false, false, false, true>
        <<<gblocks, 256, 0, stream>>>(h1b, W2P, nullptr, nullptr, Tbuf, N, R * OUTF);
    wmma_gemm_kernel<true, false, true, false>
        <<<gblocks, 256, 0, stream>>>(h1b, root2P, bias2, agg, nullptr, N, OUTF);
    scatter_kernel<64><<<eblocks, 256, 0, stream>>>(Tbuf, cntbuf, esrc, edst,
                                                    etype, agg, E, R);
    stats_kernel<<<2048, 256, 0, stream>>>(agg, stats, N, OUTF, 0);
    finalize_kernel<<<1, 128, 0, stream>>>(stats, N, OUTF);
    normalize_kernel<<<4096, 256, 0, stream>>>(agg, stats, out_final, nullptr,
                                               N, OUTF, 0);
}
